// ChunkedLSAttention_37022618091876
// MI455X (gfx1250) — matrix operand order repaired and checked by reading, not compiled, not run on hardware
//
#include <hip/hip_runtime.h>
#include <hip/hip_bf16.h>

// ---------------------------------------------------------------------------
// ChunkedLSAttention for MI455X (gfx1250).
// Heavy GEMMs (~72 GFLOP) run on v_wmma_f32_16x16x32_bf16 (wave32 WMMA,
// fp32 accumulate), register->LDS double-buffered pipeline: b128 global loads
// issue before the WMMA block, bf16 conversion + LDS stores after it.
// Attention einsums (~8.6 GFLOP) run on VALU fp32 with float4 loads; the
// q-tile is staged into LDS with GLOBAL_LOAD_ASYNC_TO_LDS_B128 (ASYNCcnt).
// ---------------------------------------------------------------------------

#define Bsz 4
#define Lq  2048
#define Dm  1024
#define Hn  16
#define DH  64
#define Wn  128
#define CSn 32
#define CACHEn 2048
#define NCCn 64        // CACHE/CS
#define NCBn 64        // L/CS
#define NCn  128       // NCC+NCB
#define NQn  64        // L/CS
#define WINROWS (Wn + Lq)   // 2176

typedef __bf16 bf16_t;
typedef __attribute__((ext_vector_type(16))) __bf16 v16bf;
typedef __attribute__((ext_vector_type(8)))  __bf16 v8bf;
typedef __attribute__((ext_vector_type(8)))  float  v8f;

// LDS byte-offset of a shared-memory pointer (AS3 ptrtoint)
__device__ inline unsigned lds_off(const void* p) {
  return (unsigned)(unsigned long long)(__attribute__((address_space(3))) const char*)p;
}

// ---------------------------------------------------------------------------
// WMMA fragment loaders (CDNA5 16-bit layouts, CDNA5 ISA §7.12.2).
// A 16x32: lanes 0-15 -> M=lane, K runs [kb..kb+7] and [kb+16..kb+23], kb=(lane>>4)*8
// B 32x16 (stored col-major in LDS): lane -> N=lane&15, K runs [kb..kb+7] and [kb+16..kb+23], kb=(lane>>4)*8
// ---------------------------------------------------------------------------
#define LDA 48   // padded row stride (halves); 96B keeps b128 loads 16B aligned
#define LDB 48

__device__ inline v16bf load_frag_a(const bf16_t* As, int m0, int lane) {
  int row = m0 + (lane & 15);
  int kb  = (lane >> 4) * 8;
  const bf16_t* p = As + row * LDA + kb;
  union { v16bf v; v8bf h[2]; } u;
  u.h[0] = *(const v8bf*)(p);
  u.h[1] = *(const v8bf*)(p + 16);
  return u.v;
}

__device__ inline v16bf load_frag_b(const bf16_t* Bs, int n0, int lane) {
  int col = n0 + (lane & 15);
  int kb  = (lane >> 4) * 8;
  const bf16_t* p = Bs + col * LDB + kb;
  union { v16bf v; v8bf h[2]; } u;
  u.h[0] = *(const v8bf*)(p);
  u.h[1] = *(const v8bf*)(p + 16);
  return u.v;
}

// ---------------------------------------------------------------------------
// Generic WMMA GEMM: C[M,N] = rowmapC( rowmapA(A)[M,K] @ B[K,N] ) + bias
// Tile 128x64x32, 256 threads (8 waves), each wave owns a 32x32 C block.
// Pipeline per k-step: issue next tile's 6 b128 global loads -> run 4 WMMAs
// on current LDS buffer -> convert/store next tile into the other buffer.
// Row maps: phys_row = (r / Lin) * Lstride + Off + (r % Lin).
// Requires M%128==0, N%64==0, K%32==0 (true for every GEMM here).
// ---------------------------------------------------------------------------
__global__ __launch_bounds__(256)
void wmma_gemm_kernel(const float* __restrict__ A, const float* __restrict__ Bw,
                      const float* __restrict__ bias, float* __restrict__ C,
                      int M, int N, int K,
                      int aLin, int aLstr, int aOff,
                      int cLin, int cLstr, int cOff)
{
  __shared__ __align__(16) bf16_t As[2][128 * LDA];
  __shared__ __align__(16) bf16_t Bs[2][64 * LDB];

  const int tid  = threadIdx.x;
  const int lane = tid & 31;
  const int wave = tid >> 5;
  const int wm   = wave & 3;   // 4 row-blocks of 32
  const int wn   = wave >> 2;  // 2 col-blocks of 32
  const int bm   = blockIdx.y * 128;
  const int bn   = blockIdx.x * 64;

  // ---- hoisted per-thread staging descriptors (k-invariant) ----
  // A tile: 128x32 f32 = 1024 float4; 4 per thread.
  const float* aPtr[4];
  int aLds[4];
  #pragma unroll
  for (int j = 0; j < 4; ++j) {
    int i  = tid + 256 * j;
    int r  = i >> 3;            // 0..127
    int c4 = (i & 7) * 4;       // 0,4,..,28
    int gr = bm + r;
    int ar = (gr / aLin) * aLstr + aOff + (gr % aLin);   // once, not per k-step
    aPtr[j] = A + (size_t)ar * K + c4;
    aLds[j] = r * LDA + c4;
  }
  // B tile: 32x64 f32 = 512 float4; 2 per thread (stored transposed in LDS).
  const float* bPtr[2];
  int bR[2], bC[2];
  #pragma unroll
  for (int j = 0; j < 2; ++j) {
    int i  = tid + 256 * j;
    int r  = i >> 4;            // k within tile: 0..31
    int c4 = (i & 15) * 4;      // n within tile: 0,4,..,60
    bPtr[j] = Bw + (size_t)r * N + bn + c4;
    bR[j] = r; bC[j] = c4;
  }

  // staging registers (distinct so the 6 b128 loads issue back-to-back)
  float4 ra0, ra1, ra2, ra3, rb0, rb1;

  auto loadRegs = [&](int k0) {
    ra0 = *(const float4*)(aPtr[0] + k0);
    ra1 = *(const float4*)(aPtr[1] + k0);
    ra2 = *(const float4*)(aPtr[2] + k0);
    ra3 = *(const float4*)(aPtr[3] + k0);
    rb0 = *(const float4*)(bPtr[0] + (size_t)k0 * N);
    rb1 = *(const float4*)(bPtr[1] + (size_t)k0 * N);
  };
  auto storeLds = [&](int buf) {
    bf16_t* d;
    d = &As[buf][aLds[0]]; d[0]=(bf16_t)ra0.x; d[1]=(bf16_t)ra0.y; d[2]=(bf16_t)ra0.z; d[3]=(bf16_t)ra0.w;
    d = &As[buf][aLds[1]]; d[0]=(bf16_t)ra1.x; d[1]=(bf16_t)ra1.y; d[2]=(bf16_t)ra1.z; d[3]=(bf16_t)ra1.w;
    d = &As[buf][aLds[2]]; d[0]=(bf16_t)ra2.x; d[1]=(bf16_t)ra2.y; d[2]=(bf16_t)ra2.z; d[3]=(bf16_t)ra2.w;
    d = &As[buf][aLds[3]]; d[0]=(bf16_t)ra3.x; d[1]=(bf16_t)ra3.y; d[2]=(bf16_t)ra3.z; d[3]=(bf16_t)ra3.w;
    bf16_t* base = Bs[buf];
    base[(bC[0]+0)*LDB + bR[0]] = (bf16_t)rb0.x;
    base[(bC[0]+1)*LDB + bR[0]] = (bf16_t)rb0.y;
    base[(bC[0]+2)*LDB + bR[0]] = (bf16_t)rb0.z;
    base[(bC[0]+3)*LDB + bR[0]] = (bf16_t)rb0.w;
    base[(bC[1]+0)*LDB + bR[1]] = (bf16_t)rb1.x;
    base[(bC[1]+1)*LDB + bR[1]] = (bf16_t)rb1.y;
    base[(bC[1]+2)*LDB + bR[1]] = (bf16_t)rb1.z;
    base[(bC[1]+3)*LDB + bR[1]] = (bf16_t)rb1.w;
  };

  v8f acc00 = {}, acc01 = {}, acc10 = {}, acc11 = {};

  loadRegs(0);
  storeLds(0);
  __syncthreads();

  const int nk = K >> 5;
  for (int t = 0; t < nk; ++t) {
    const int buf = t & 1;
    if (t + 1 < nk) loadRegs((t + 1) << 5);   // global loads fly over the WMMAs

    v16bf a0 = load_frag_a(As[buf], wm * 32,      lane);
    v16bf a1 = load_frag_a(As[buf], wm * 32 + 16, lane);
    v16bf b0 = load_frag_b(Bs[buf], wn * 32,      lane);
    v16bf b1 = load_frag_b(Bs[buf], wn * 32 + 16, lane);

    acc00 = __builtin_amdgcn_wmma_f32_16x16x32_bf16(false, a0, false, b0, (short)0, acc00, false, false);
    acc01 = __builtin_amdgcn_wmma_f32_16x16x32_bf16(false, a0, false, b1, (short)0, acc01, false, false);
    acc10 = __builtin_amdgcn_wmma_f32_16x16x32_bf16(false, a1, false, b0, (short)0, acc10, false, false);
    acc11 = __builtin_amdgcn_wmma_f32_16x16x32_bf16(false, a1, false, b1, (short)0, acc11, false, false);

    if (t + 1 < nk) storeLds(buf ^ 1);        // wait-loadcnt lands here, post-WMMA
    __syncthreads();
  }

  // C layout: lane -> N = n0+(lane&15); VGPR r -> M = m0 + r + 8*(lane>>4)
  const int mb = (lane >> 4) << 3;
  const int nl = lane & 15;
  #pragma unroll
  for (int ij = 0; ij < 4; ++ij) {
    int i = ij >> 1, j = ij & 1;
    const v8f& acc = (ij == 0) ? acc00 : (ij == 1) ? acc01 : (ij == 2) ? acc10 : acc11;
    int m0  = bm + wm * 32 + i * 16;
    int n0  = bn + wn * 32 + j * 16;
    int col = n0 + nl;
    float bv = bias ? bias[col] : 0.0f;
    #pragma unroll
    for (int r = 0; r < 8; ++r) {
      int m  = m0 + mb + r;
      int cr = (m / cLin) * cLstr + cOff + (m % cLin);
      C[(size_t)cr * N + col] = acc[r] + bv;
    }
  }
}

// ---------------------------------------------------------------------------
// Compression scores: S[b][h][chunk][s] = dot(X[b, chunk*CS+s], Wd[:,h]) + bd[h]
// ---------------------------------------------------------------------------
__global__ void scores_raw_kernel(const float* __restrict__ X, const float* __restrict__ Wd,
                                  const float* __restrict__ bd, float* __restrict__ S,
                                  int rowsPerBatch, int nc)
{
  int idx = blockIdx.x * 256 + threadIdx.x;
  int total = Bsz * nc * CSn * Hn;
  if (idx >= total) return;
  int h = idx & 15;
  int t = idx >> 4;
  int s = t & 31;  t >>= 5;
  int c = t % nc;
  int b = t / nc;
  const float* x = X + ((size_t)b * rowsPerBatch + c * CSn + s) * Dm;
  float acc = bd[h];
  #pragma unroll 8
  for (int d = 0; d < Dm; ++d) acc = fmaf(x[d], Wd[d * Hn + h], acc);
  S[(((size_t)(b * Hn + h) * nc + c) * CSn) + s] = acc;
}

__global__ void softmax_cs_kernel(float* __restrict__ S, int n)
{
  int i = blockIdx.x * 256 + threadIdx.x;
  if (i >= n) return;
  float* p = S + (size_t)i * CSn;
  float mx = p[0];
  for (int s = 1; s < CSn; ++s) mx = fmaxf(mx, p[s]);
  float sum = 0.f;
  for (int s = 0; s < CSn; ++s) { float e = __expf(p[s] - mx); p[s] = e; sum += e; }
  float inv = 1.0f / sum;
  for (int s = 0; s < CSn; ++s) p[s] *= inv;
}

// ---------------------------------------------------------------------------
// Chunk compression: OUT[(b*nc+c), h*64+dh] = sum_s SC[b,h,c,s] * X[row(c,s), h*64+dh]
// ---------------------------------------------------------------------------
__global__ void compress_kernel(const float* __restrict__ X, const float* __restrict__ SC,
                                float* __restrict__ OUT, int rowsPerBatch, int rowOff, int nc)
{
  int idx = blockIdx.x * 256 + threadIdx.x;
  int total = Bsz * nc * Dm;
  if (idx >= total) return;
  int d = idx & (Dm - 1);
  int t = idx >> 10;
  int c = t % nc;
  int b = t / nc;
  int h = d >> 6;
  const float* scp = SC + (((size_t)(b * Hn + h) * nc + c) * CSn);
  const float* xp  = X + ((size_t)b * rowsPerBatch + rowOff + c * CSn) * Dm + d;
  float acc = 0.f;
  #pragma unroll
  for (int s = 0; s < CSn; ++s) acc = fmaf(scp[s], xp[(size_t)s * Dm], acc);
  OUT[(size_t)(b * nc + c) * Dm + d] = acc;
}

// ---------------------------------------------------------------------------
// Row LayerNorm over D=1024 (in place), one 256-thread WG per row.
// ---------------------------------------------------------------------------
__global__ __launch_bounds__(256)
void ln_kernel(float* __restrict__ X, const float* __restrict__ g, const float* __restrict__ be)
{
  float* x = X + (size_t)blockIdx.x * Dm;
  __shared__ float red[256];
  int tid = threadIdx.x;
  float s = 0.f;
  for (int d = tid; d < Dm; d += 256) s += x[d];
  red[tid] = s; __syncthreads();
  for (int o = 128; o > 0; o >>= 1) { if (tid < o) red[tid] += red[tid + o]; __syncthreads(); }
  float mean = red[0] * (1.0f / Dm);
  __syncthreads();
  float v = 0.f;
  for (int d = tid; d < Dm; d += 256) { float t = x[d] - mean; v = fmaf(t, t, v); }
  red[tid] = v; __syncthreads();
  for (int o = 128; o > 0; o >>= 1) { if (tid < o) red[tid] += red[tid + o]; __syncthreads(); }
  float rstd = rsqrtf(red[0] * (1.0f / Dm) + 1e-5f);
  __syncthreads();
  for (int d = tid; d < Dm; d += 256) x[d] = (x[d] - mean) * rstd * g[d] + be[d];
}

// ---------------------------------------------------------------------------
// q -> head layout with 1/sqrt(DH) scale: qh[b,h,l,d] = IN[b,l,h*64+d] * 0.125
// ---------------------------------------------------------------------------
__global__ void qheads_kernel(const float* __restrict__ IN, float* __restrict__ qh)
{
  int idx = blockIdx.x * 256 + threadIdx.x;   // B*L*D
  if (idx >= Bsz * Lq * Dm) return;
  int d = idx & (Dm - 1);
  int l = (idx >> 10) & (Lq - 1);
  int b = idx >> 21;
  int h = d >> 6;
  int dd = d & 63;
  qh[(((size_t)(b * Hn + h)) * Lq + l) * DH + dd] = IN[idx] * 0.125f;
}

// k_comp/v_comp = concat([cache_rows, bp_rows], chunk) in head layout
__global__ void compheads_kernel(const float* __restrict__ CROWS, const float* __restrict__ BROWS,
                                 float* __restrict__ OUT)
{
  int idx = blockIdx.x * 256 + threadIdx.x;   // B*H*NC*DH
  if (idx >= Bsz * Hn * NCn * DH) return;
  int d = idx & 63;
  int c = (idx >> 6) & (NCn - 1);
  int h = (idx >> 13) & (Hn - 1);
  int b = idx >> 17;
  float v;
  if (c < NCCn) v = CROWS[(size_t)(b * NCCn + c) * Dm + h * DH + d];
  else          v = BROWS[(size_t)(b * NCBn + (c - NCCn)) * Dm + h * DH + d];
  OUT[idx] = v;
}

// ---------------------------------------------------------------------------
// Fused attention: one WG per (b, h, 16-query tile).
// 256 scores/query: [0,128) compressed keys (+ analytic rel-shift pos gather,
// causal chunk mask), [128,256) sliding window of W causal keys (+ window pos).
// q-tile staged via GLOBAL_LOAD_ASYNC_TO_LDS_B128; softmax in LDS; context
// accumulated straight into [B,L,D] layout.
// ---------------------------------------------------------------------------
__device__ inline float dot64(const float* __restrict__ a, const float* __restrict__ b)
{
  const float4* a4 = (const float4*)a;
  const float4* b4 = (const float4*)b;
  float s = 0.f;
  #pragma unroll
  for (int i = 0; i < 16; ++i) {
    float4 x = a4[i], y = b4[i];
    s = fmaf(x.x, y.x, s); s = fmaf(x.y, y.y, s);
    s = fmaf(x.z, y.z, s); s = fmaf(x.w, y.w, s);
  }
  return s;
}

__global__ __launch_bounds__(256)
void attn_kernel(const float* __restrict__ qh,   const float* __restrict__ kcomp,
                 const float* __restrict__ vcomp, const float* __restrict__ kwin,
                 const float* __restrict__ vwin,  const float* __restrict__ post,
                 const float* __restrict__ posc,  const float* __restrict__ rrb,
                 const float* __restrict__ rwb,   float* __restrict__ ctx)
{
  const int blk  = blockIdx.x;            // (b*H + h)*(L/16) + tile
  const int tile = blk & 127;             // L/16 = 128 tiles
  const int bh   = blk >> 7;
  const int h    = bh & (Hn - 1);
  const int b    = bh >> 4;
  const int l0   = tile * 16;

  __shared__ __align__(16) float qs[16][DH];
  __shared__ __align__(16) float qrw[16][DH];
  __shared__ __align__(16) float qrr[16][DH];
  __shared__ float sc[16][256];

  const int tid = threadIdx.x;

  // ---- async global->LDS stage of the 16x64 q tile (one b128 per lane) ----
  {
    int r  = tid >> 4;              // 0..15
    int d4 = (tid & 15) * 4;        // 0,4,..,60
    const float* gp = qh + ((size_t)bh * Lq + l0 + r) * DH + d4;
    unsigned long long ga = (unsigned long long)gp;
    unsigned loff = lds_off(&qs[r][d4]);
    asm volatile("global_load_async_to_lds_b128 %0, %1, off"
                 :: "v"(loff), "v"(ga) : "memory");
    asm volatile("s_wait_asynccnt 0x0" ::: "memory");
  }
  __syncthreads();
  for (int i = tid; i < 16 * DH; i += 256) {
    int r = i >> 6, d = i & 63;
    float qv = qs[r][d];
    qrw[r][d] = qv + rwb[h * DH + d];   // q + r_w_bias (window keys)
    qrr[r][d] = qv + rrb[h * DH + d];   // q + r_r_bias (window pos)
  }
  __syncthreads();

  for (int i = tid; i < 16 * 256; i += 256) {
    int r = i >> 8, c = i & 255;
    int l = l0 + r;
    float s;
    if (c < NCn) {
      s = dot64(qs[r], kcomp + ((size_t)bh * NCn + c) * DH);
      // rel_shift: f = NQ + qchunk*NC + c ; q' = f/129, c' = f%129 (c'==NC => 0)
      int f  = NQn + (l >> 5) * NCn + c;
      int qp = f / (NCn + 1);
      int cp = f - qp * (NCn + 1);
      if (cp < NCn) {
        int lp = qp * CSn + (l & 31);
        s += dot64(qh + ((size_t)bh * Lq + lp) * DH, posc + (size_t)cp * Dm + h * DH);
      }
      if (c >= NCCn && (c - NCCn) >= (l >> 5)) s = -1e30f;   // causal chunk mask
    } else {
      int w  = c - NCn;
      int kr = b * WINROWS + l + 1 + w;   // causal window keys (win coords)
      s = dot64(qrw[r], kwin + (size_t)kr * Dm + h * DH)
        + dot64(qrr[r], post + (size_t)w * Dm + h * DH);
    }
    sc[r][c] = s;
  }
  __syncthreads();

  if (tid < 16) {
    float mx = -3.4e38f;
    for (int c = 0; c < 256; ++c) mx = fmaxf(mx, sc[tid][c]);
    float sum = 0.f;
    for (int c = 0; c < 256; ++c) { float e = __expf(sc[tid][c] - mx); sc[tid][c] = e; sum += e; }
    float inv = 1.0f / sum;
    for (int c = 0; c < 256; ++c) sc[tid][c] *= inv;
  }
  __syncthreads();

  for (int i = tid; i < 16 * DH; i += 256) {
    int r = i >> 6, d = i & 63;
    int l = l0 + r;
    float acc = 0.f;
    const float* vcp = vcomp + (size_t)bh * NCn * DH + d;
    #pragma unroll 4
    for (int c = 0; c < NCn; ++c) acc = fmaf(sc[r][c], vcp[c * DH], acc);
    const float* vwp = vwin + ((size_t)b * WINROWS + l + 1) * Dm + h * DH + d;
    #pragma unroll 4
    for (int w = 0; w < Wn; ++w) acc = fmaf(sc[r][NCn + w], vwp[(size_t)w * Dm], acc);
    ctx[((size_t)b * Lq + l) * Dm + h * DH + d] = acc;
  }
}

// ---------------------------------------------------------------------------
// Host orchestration
// ---------------------------------------------------------------------------
extern "C" void kernel_launch(void* const* d_in, const int* in_sizes, int n_in,
                              void* d_out, int out_size, void* d_ws, size_t ws_size,
                              hipStream_t stream)
{
  const float* h       = (const float*)d_in[0];
  const float* h_cache = (const float*)d_in[1];
  const float* key_pe  = (const float*)d_in[2];
  const float* pos_win = (const float*)d_in[3];
  const float* Wq  = (const float*)d_in[4];
  const float* Wk  = (const float*)d_in[5];
  const float* Wv  = (const float*)d_in[6];
  const float* Wo  = (const float*)d_in[7];
  const float* Wd  = (const float*)d_in[8];
  const float* bd  = (const float*)d_in[9];
  const float* Wr  = (const float*)d_in[10];
  const float* br  = (const float*)d_in[11];
  const float* Wrc = (const float*)d_in[12];
  const float* brc = (const float*)d_in[13];
  const float* rrb = (const float*)d_in[14];   // r_r_bias [1,H,1,DH]
  const float* rwb = (const float*)d_in[15];   // r_w_bias [1,H,1,1,DH]
  const float* g_dp = (const float*)d_in[16];
  const float* b_dp = (const float*)d_in[17];
  const float* g_w  = (const float*)d_in[18];
  const float* b_w  = (const float*)d_in[19];
  // d_in[20] = chunked_attn_mask: recomputed analytically in attn_kernel.

  // workspace carve-out (fp32), ~146 MB total
  float* w = (float*)d_ws;
  size_t off = 0;
  auto alloc = [&](size_t n) { float* p = w + off; off += n; return p; };
  float* qh    = alloc((size_t)Bsz * Hn * Lq * DH);        // 8.39M
  float* k_win = alloc((size_t)Bsz * WINROWS * Dm);        // 8.91M
  float* v_win = alloc((size_t)Bsz * WINROWS * Dm);        // 8.91M
  float* ctx   = alloc((size_t)Bsz * Lq * Dm);             // 8.39M
  float* hcm   = alloc((size_t)Bsz * NCCn * Dm);
  float* kc    = alloc((size_t)Bsz * NCCn * Dm);
  float* vc    = alloc((size_t)Bsz * NCCn * Dm);
  float* kbpc  = alloc((size_t)Bsz * NCBn * Dm);
  float* vbpc  = alloc((size_t)Bsz * NCBn * Dm);
  float* sc_c  = alloc((size_t)Bsz * Hn * NCCn * CSn);
  float* sc_h  = alloc((size_t)Bsz * Hn * NCBn * CSn);
  float* posT  = alloc((size_t)Wn * Dm);
  float* posC  = alloc((size_t)NCn * Dm);
  float* kcomp = alloc((size_t)Bsz * Hn * NCn * DH);
  float* vcomp = alloc((size_t)Bsz * Hn * NCn * DH);
  (void)ws_size; (void)n_in; (void)in_sizes; (void)out_size;

  auto gemm = [&](const float* A, const float* Bm, const float* bias, float* C,
                  int M, int N, int K, int aLin, int aLstr, int aOff,
                  int cLin, int cLstr, int cOff) {
    dim3 g(N / 64, M / 128);
    wmma_gemm_kernel<<<g, 256, 0, stream>>>(A, Bm, bias, C, M, N, K,
                                            aLin, aLstr, aOff, cLin, cLstr, cOff);
  };

  // 1) Q = h@Wq -> heads (scaled)
  gemm(h, Wq, nullptr, ctx, Bsz * Lq, Dm, Dm, Bsz * Lq, Bsz * Lq, 0, Bsz * Lq, Bsz * Lq, 0);
  qheads_kernel<<<(Bsz * Lq * Dm) / 256, 256, 0, stream>>>(ctx, qh);

  // 2) k_bp/v_bp = h@Wk / h@Wv written straight into win buffers rows [W, W+L)
  gemm(h, Wk, nullptr, k_win, Bsz * Lq, Dm, Dm, Lq, Lq, 0, Lq, WINROWS, Wn);
  gemm(h, Wv, nullptr, v_win, Bsz * Lq, Dm, Dm, Lq, Lq, 0, Lq, WINROWS, Wn);

  // 3) cache tail hw = h_cache[:, -W:] projected into win rows [0, W)
  gemm(h_cache, Wk, nullptr, k_win, Bsz * Wn, Dm, Dm, Wn, CACHEn, CACHEn - Wn, Wn, WINROWS, 0);
  gemm(h_cache, Wv, nullptr, v_win, Bsz * Wn, Dm, Dm, Wn, CACHEn, CACHEn - Wn, Wn, WINROWS, 0);

  // 4) cache compression scores + compress -> hcm rows [B*NCC, D]
  scores_raw_kernel<<<(Bsz * NCCn * CSn * Hn) / 256, 256, 0, stream>>>(h_cache, Wd, bd, sc_c, CACHEn, NCCn);
  softmax_cs_kernel<<<(Bsz * Hn * NCCn) / 256, 256, 0, stream>>>(sc_c, Bsz * Hn * NCCn);
  compress_kernel<<<(Bsz * NCCn * Dm) / 256, 256, 0, stream>>>(h_cache, sc_c, hcm, CACHEn, 0, NCCn);

  // 5) k_cache/v_cache rows = hcm @ Wk/Wv
  gemm(hcm, Wk, nullptr, kc, Bsz * NCCn, Dm, Dm, Bsz * NCCn, Bsz * NCCn, 0, Bsz * NCCn, Bsz * NCCn, 0);
  gemm(hcm, Wv, nullptr, vc, Bsz * NCCn, Dm, Dm, Bsz * NCCn, Bsz * NCCn, 0, Bsz * NCCn, Bsz * NCCn, 0);

  // 6) h compression scores; compress k_bp/v_bp (pre-LN win rows W..)
  scores_raw_kernel<<<(Bsz * NCBn * CSn * Hn) / 256, 256, 0, stream>>>(h, Wd, bd, sc_h, Lq, NCBn);
  softmax_cs_kernel<<<(Bsz * Hn * NCBn) / 256, 256, 0, stream>>>(sc_h, Bsz * Hn * NCBn);
  compress_kernel<<<(Bsz * NCBn * Dm) / 256, 256, 0, stream>>>(k_win, sc_h, kbpc, WINROWS, Wn, NCBn);
  compress_kernel<<<(Bsz * NCBn * Dm) / 256, 256, 0, stream>>>(v_win, sc_h, vbpc, WINROWS, Wn, NCBn);

  // 7) LayerNorms (in place; win LN AFTER compression reads the raw rows)
  ln_kernel<<<Bsz * WINROWS, 256, 0, stream>>>(k_win, g_w, b_w);
  ln_kernel<<<Bsz * WINROWS, 256, 0, stream>>>(v_win, g_w, b_w);
  ln_kernel<<<Bsz * NCCn, 256, 0, stream>>>(kc,   g_dp, b_dp);
  ln_kernel<<<Bsz * NCCn, 256, 0, stream>>>(vc,   g_dp, b_dp);
  ln_kernel<<<Bsz * NCBn, 256, 0, stream>>>(kbpc, g_dp, b_dp);
  ln_kernel<<<Bsz * NCBn, 256, 0, stream>>>(vbpc, g_dp, b_dp);

  // 8) pack compressed K/V into head layout
  compheads_kernel<<<(Bsz * Hn * NCn * DH) / 256, 256, 0, stream>>>(kc, kbpc, kcomp);
  compheads_kernel<<<(Bsz * Hn * NCn * DH) / 256, 256, 0, stream>>>(vc, vbpc, vcomp);

  // 9) positional projections (with bias)
  gemm(pos_win, Wr,  br,  posT, Wn,  Dm, Dm, Wn,  Wn,  0, Wn,  Wn,  0);
  gemm(key_pe,  Wrc, brc, posC, NCn, Dm, Dm, NCn, NCn, 0, NCn, NCn, 0);

  // 10) fused attention -> ctx [B,L,D]
  attn_kernel<<<Bsz * Hn * (Lq / 16), 256, 0, stream>>>(qh, kcomp, vcomp, k_win, v_win,
                                                        posT, posC, rrb, rwb, ctx);

  // 11) output projection
  gemm(ctx, Wo, nullptr, (float*)d_out, Bsz * Lq, Dm, Dm,
       Bsz * Lq, Bsz * Lq, 0, Bsz * Lq, Bsz * Lq, 0);
}
